// TTTWrapper_71614284694326
// MI455X (gfx1250) — compile-verified, matrix operand order checked by reading, not run on hardware
//
#include <hip/hip_runtime.h>

typedef _Float16 half_t;
typedef half_t v16h __attribute__((ext_vector_type(16)));
typedef half_t v8h  __attribute__((ext_vector_type(8)));
typedef half_t v4h  __attribute__((ext_vector_type(4)));
typedef float  v8f  __attribute__((ext_vector_type(8)));
typedef float  v4f  __attribute__((ext_vector_type(4)));

#define NORM_EPS   1e-6f
#define G_REL_CAP  0.02f
#define DW_REL_CAP 0.1f

// Scalar slot layout in workspace (float array "scal"):
//   scal[0]                 : ||W0||_F^2
//   scal[1  + 2*c + b]      : ||G_c||_F^2 (unprojected), c = chunk
//   scal[33 + 2*c + b]      : ||dW||_F^2 before projection entering chunk c
//                             (slot c=0 stays zero: initial dW = 0)

__device__ inline float waveReduceSum(float v) {
#pragma unroll
  for (int o = 16; o > 0; o >>= 1) v += __shfl_down(v, o, 32);
  return v;
}

__device__ inline v8h ldh8(const half_t* p) { return *(const v8h*)p; }

__device__ inline v16h cat16(v8h lo, v8h hi) {
  return __builtin_shufflevector(lo, hi, 0, 1, 2, 3, 4, 5, 6, 7, 8, 9, 10, 11,
                                 12, 13, 14, 15);
}

// ---------------------------------------------------------------------------
// ||W0||^2 reduction
__global__ __launch_bounds__(256)
void sumsq_kernel(const float* __restrict__ x, long n4, float* __restrict__ dst) {
  float s = 0.f;
  const long stride = (long)gridDim.x * blockDim.x;
  const v4f* x4 = (const v4f*)x;
  for (long i = (long)blockIdx.x * blockDim.x + threadIdx.x; i < n4; i += stride) {
    v4f f = x4[i];
    s += f[0] * f[0] + f[1] * f[1] + f[2] * f[2] + f[3] * f[3];
  }
  s = waveReduceSum(s);
  if ((threadIdx.x & 31) == 0) atomicAdd(dst, s);
}

// ---------------------------------------------------------------------------
// per-token 1/rms (one wave per row; rowLen multiple of 128)
__global__ __launch_bounds__(256)
void row_rms_kernel(const float* __restrict__ x, float* __restrict__ out,
                    int rowLen, long nRows) {
  const int lane = threadIdx.x & 31;
  const long wave = ((long)blockIdx.x * blockDim.x + threadIdx.x) >> 5;
  if (wave >= nRows) return;
  const float* row = x + wave * (long)rowLen;
  float s = 0.f;
  for (int i = lane * 4; i < rowLen; i += 128) {
    v4f f = *(const v4f*)(row + i);
    s += f[0] * f[0] + f[1] * f[1] + f[2] * f[2] + f[3] * f[3];
  }
  s = waveReduceSum(s);
  if (lane == 0) out[wave] = rsqrtf(s / (float)rowLen + NORM_EPS);
}

// ---------------------------------------------------------------------------
// Bulk f32 -> f16 conversion (z staged once for the whole sequence)
__global__ __launch_bounds__(256)
void cvt_f16_kernel(const float* __restrict__ src, half_t* __restrict__ dst,
                    long n8) {
  const long stride = (long)gridDim.x * blockDim.x;
  for (long i = (long)blockIdx.x * blockDim.x + threadIdx.x; i < n8; i += stride) {
    const long i8 = i << 3;
    v4f a = *(const v4f*)(src + i8);
    v4f b = *(const v4f*)(src + i8 + 4);
    v8h h;
#pragma unroll
    for (int j = 0; j < 4; ++j) {
      h[j] = (half_t)a[j];
      h[4 + j] = (half_t)b[j];
    }
    *(v8h*)(dst + i8) = h;
  }
}

// ---------------------------------------------------------------------------
// Per-chunk: Weff = f16(W0 + s_k * dW), s_k = Frobenius projection scale.
__global__ __launch_bounds__(256)
void prep_weff_kernel(const float* __restrict__ W0, const float* __restrict__ dW,
                      const float* __restrict__ scal, half_t* __restrict__ Weff,
                      int chunkIdx, long perBatch, int B) {
  const float w0n = sqrtf(scal[0]);
  const long n8 = (long)B * (perBatch >> 3);
  const long stride = (long)gridDim.x * blockDim.x;
  for (long i = (long)blockIdx.x * blockDim.x + threadIdx.x; i < n8; i += stride) {
    const long i8 = i << 3;
    const int b = (int)(i8 / perBatch);
    const long r = i8 - (long)b * perBatch;
    const long ip8 = (i + stride) << 3;
    if (ip8 < (long)B * perBatch) {   // stream-ahead prefetch
      __builtin_prefetch(dW + ip8, 0, 1);
      __builtin_prefetch(W0 + (ip8 >= perBatch ? ip8 - perBatch : ip8), 0, 1);
    }
    const float dwn2 = scal[33 + 2 * chunkIdx + b];
    const float s = fminf(DW_REL_CAP * w0n / (sqrtf(dwn2) + 1e-8f), 1.0f);
    v4f wa = *(const v4f*)(W0 + r);
    v4f wb = *(const v4f*)(W0 + r + 4);
    v4f da = *(const v4f*)(dW + i8);
    v4f db = *(const v4f*)(dW + i8 + 4);
    v8h h;
#pragma unroll
    for (int j = 0; j < 4; ++j) {
      h[j] = (half_t)(wa[j] + s * da[j]);
      h[4 + j] = (half_t)(wb[j] + s * db[j]);
    }
    *(v8h*)(Weff + i8) = h;
  }
}

// ---------------------------------------------------------------------------
// Per-chunk: dst[b][d][C] = f16(src[b][t0+c][d] * rms[b][t0+c]) transposed,
// so grad-GEMM operands become K(=c)-contiguous.  64x64 LDS tile transpose.
__global__ __launch_bounds__(256)
void transpose_norm_kernel(const float* __restrict__ src,
                           const float* __restrict__ rms,
                           half_t* __restrict__ dst, int T, int d,
                           int chunkIdx) {
  const int C = 256;
  __shared__ half_t tile[64][72];  // [d_local][c_local], padded
  const int dTiles = d >> 6;
  const int bid = blockIdx.x;
  const int b = bid / (4 * dTiles);
  const int rem = bid - b * 4 * dTiles;
  const int ct = rem / dTiles;
  const int dt = rem - ct * dTiles;
  const long t0 = (long)chunkIdx * C + ct * 64;
  const int d0 = dt << 6;
  const int tx = threadIdx.x & 15;
  const int ty = threadIdx.x >> 4;
#pragma unroll
  for (int rr = 0; rr < 4; ++rr) {
    const int c = ty + rr * 16;
    const long trow = (long)b * T + t0 + c;
    const float r = rms[trow];
    v4f f = *(const v4f*)(src + trow * d + d0 + tx * 4);
#pragma unroll
    for (int j = 0; j < 4; ++j) tile[tx * 4 + j][c] = (half_t)(f[j] * r);
  }
  __syncthreads();
#pragma unroll
  for (int rr = 0; rr < 4; ++rr) {
    const int dl = ty + rr * 16;
    v4h h = *(const v4h*)&tile[dl][tx * 4];
    *(v4h*)(dst + ((long)b * d + d0 + dl) * C + ct * 64 + tx * 4) = h;
  }
}

// ---------------------------------------------------------------------------
// Apply: o = z_chunk @ Weff^T + bias.  Wave -> 16 rows x 64 cols.
// Block's 8 waves share the same 64 Weff rows (cache locality).
__global__ __launch_bounds__(256)
void ttt_apply(const half_t* __restrict__ zf16, const half_t* __restrict__ Weff,
               const float* __restrict__ bias, float* __restrict__ out,
               int chunkIdx, int T, int d_in, int d_out) {
  const int C = 256;
  const int lane = threadIdx.x & 31;
  const int wave = (int)(((long)blockIdx.x * blockDim.x + threadIdx.x) >> 5);
  const int colGroups = d_out >> 6;
  const int rowTiles = C >> 4;  // 16
  const int wavesPerBatch = colGroups * rowTiles;
  const int b = wave / wavesPerBatch;
  const int rem = wave - b * wavesPerBatch;
  const int rowTile = rem & (rowTiles - 1);  // fast-varying: block shares colGrp
  const int colGrp = rem / rowTiles;

  const int m = lane & 15;
  const int kg = (lane >> 4) << 3;
  const long t0 = (long)chunkIdx * C + rowTile * 16;

  const half_t* arow = zf16 + ((long)b * T + t0 + m) * d_in + kg;
  const half_t* brow0 = Weff + ((long)b * d_out + ((colGrp << 6) + m)) * d_in + kg;
  const long bstride = (long)16 * d_in;

  v8f acc[4] = {};
  v8h aA0 = ldh8(arow), aA1 = ldh8(arow + 16);
  v8h bA0[4], bA1[4];
#pragma unroll
  for (int nt = 0; nt < 4; ++nt) {
    bA0[nt] = ldh8(brow0 + nt * bstride);
    bA1[nt] = ldh8(brow0 + nt * bstride + 16);
  }
  for (int k = 0; k < d_in; k += 64) {
    v8h aB0 = ldh8(arow + k + 32), aB1 = ldh8(arow + k + 48);
    v8h bB0[4], bB1[4];
#pragma unroll
    for (int nt = 0; nt < 4; ++nt) {
      bB0[nt] = ldh8(brow0 + nt * bstride + k + 32);
      bB1[nt] = ldh8(brow0 + nt * bstride + k + 48);
    }
    {
      const v16h a = cat16(aA0, aA1);
#pragma unroll
      for (int nt = 0; nt < 4; ++nt)
        acc[nt] = __builtin_amdgcn_wmma_f32_16x16x32_f16(
            false, a, false, cat16(bA0[nt], bA1[nt]), (short)0, acc[nt], false,
            false);
    }
    if (k + 64 < d_in) {
      aA0 = ldh8(arow + k + 64);
      aA1 = ldh8(arow + k + 80);
#pragma unroll
      for (int nt = 0; nt < 4; ++nt) {
        bA0[nt] = ldh8(brow0 + nt * bstride + k + 64);
        bA1[nt] = ldh8(brow0 + nt * bstride + k + 80);
      }
    }
    {
      const v16h a = cat16(aB0, aB1);
#pragma unroll
      for (int nt = 0; nt < 4; ++nt)
        acc[nt] = __builtin_amdgcn_wmma_f32_16x16x32_f16(
            false, a, false, cat16(bB0[nt], bB1[nt]), (short)0, acc[nt], false,
            false);
    }
  }

  const int hi = (lane >> 4) << 3;
#pragma unroll
  for (int nt = 0; nt < 4; ++nt) {
    const int n = (colGrp << 6) + nt * 16 + (lane & 15);
    const float bv = bias[n];
#pragma unroll
    for (int r = 0; r < 8; ++r) {
      const long row = (long)b * T + t0 + r + hi;
      out[row * d_out + n] = acc[nt][r] + bv;
    }
  }
}

// ---------------------------------------------------------------------------
// Gradient: G[o,i] = (1/C) sum_c vn[c,o]*zn[c,i].  32x64 register tile:
// 8 WMMA per 12 b128 loads per K-step.  Wave -> 32 x 64 tile, double-buffered.
__global__ __launch_bounds__(256)
void ttt_grad(const half_t* __restrict__ znT, const half_t* __restrict__ vnT,
              float* __restrict__ G, float* __restrict__ scal, int chunkIdx,
              int d_in, int d_out) {
  const int C = 256;
  const int lane = threadIdx.x & 31;
  const int wave = (int)(((long)blockIdx.x * blockDim.x + threadIdx.x) >> 5);
  const int oTiles = d_out >> 5;   // 32-row tiles over d_out
  const int iGroups = d_in >> 6;   // 64-col groups over d_in
  const int wavesPerBatch = oTiles * iGroups;
  const int b = wave / wavesPerBatch;
  const int rem = wave - b * wavesPerBatch;
  const int oTile = rem % oTiles;  // fast-varying: block shares znT tile
  const int iGrp = rem / oTiles;

  const int nm = lane & 15;
  const int kg = (lane >> 4) << 3;

  const half_t* arow0 = vnT + ((long)b * d_out + oTile * 32 + nm) * C + kg;
  const half_t* arow1 = arow0 + (long)16 * C;
  const half_t* brow0 = znT + ((long)b * d_in + ((iGrp << 6) + nm)) * C + kg;
  const long bstr = (long)16 * C;

  v8f acc[2][4] = {};
  v8h aA[2][2], bA[4][2];
  aA[0][0] = ldh8(arow0);
  aA[0][1] = ldh8(arow0 + 16);
  aA[1][0] = ldh8(arow1);
  aA[1][1] = ldh8(arow1 + 16);
#pragma unroll
  for (int nt = 0; nt < 4; ++nt) {
    bA[nt][0] = ldh8(brow0 + nt * bstr);
    bA[nt][1] = ldh8(brow0 + nt * bstr + 16);
  }
  for (int k = 0; k < C; k += 64) {
    v8h aB[2][2], bB[4][2];
    aB[0][0] = ldh8(arow0 + k + 32);
    aB[0][1] = ldh8(arow0 + k + 48);
    aB[1][0] = ldh8(arow1 + k + 32);
    aB[1][1] = ldh8(arow1 + k + 48);
#pragma unroll
    for (int nt = 0; nt < 4; ++nt) {
      bB[nt][0] = ldh8(brow0 + nt * bstr + k + 32);
      bB[nt][1] = ldh8(brow0 + nt * bstr + k + 48);
    }
#pragma unroll
    for (int mt = 0; mt < 2; ++mt) {
      const v16h a = cat16(aA[mt][0], aA[mt][1]);
#pragma unroll
      for (int nt = 0; nt < 4; ++nt)
        acc[mt][nt] = __builtin_amdgcn_wmma_f32_16x16x32_f16(
            false, a, false, cat16(bA[nt][0], bA[nt][1]), (short)0, acc[mt][nt],
            false, false);
    }
    if (k + 64 < C) {
      aA[0][0] = ldh8(arow0 + k + 64);
      aA[0][1] = ldh8(arow0 + k + 80);
      aA[1][0] = ldh8(arow1 + k + 64);
      aA[1][1] = ldh8(arow1 + k + 80);
#pragma unroll
      for (int nt = 0; nt < 4; ++nt) {
        bA[nt][0] = ldh8(brow0 + nt * bstr + k + 64);
        bA[nt][1] = ldh8(brow0 + nt * bstr + k + 80);
      }
    }
#pragma unroll
    for (int mt = 0; mt < 2; ++mt) {
      const v16h a = cat16(aB[mt][0], aB[mt][1]);
#pragma unroll
      for (int nt = 0; nt < 4; ++nt)
        acc[mt][nt] = __builtin_amdgcn_wmma_f32_16x16x32_f16(
            false, a, false, cat16(bB[nt][0], bB[nt][1]), (short)0, acc[mt][nt],
            false, false);
    }
  }

  const float inv = 1.0f / (float)C;
  const int hi = (lane >> 4) << 3;
  float ss = 0.f;
#pragma unroll
  for (int mt = 0; mt < 2; ++mt) {
#pragma unroll
    for (int nt = 0; nt < 4; ++nt) {
      const long icol = (long)(iGrp << 6) + nt * 16 + nm;
#pragma unroll
      for (int r = 0; r < 8; ++r) {
        const float g = acc[mt][nt][r] * inv;
        ss += g * g;
        G[((long)b * d_out + oTile * 32 + mt * 16 + r + hi) * d_in + icol] = g;
      }
    }
  }
  ss = waveReduceSum(ss);
  if (lane == 0) atomicAdd(&scal[1 + 2 * chunkIdx + b], ss);
}

// ---------------------------------------------------------------------------
// EMA + projections: dW_new = decay*(s_prev*dW) + (1-decay)*eta*(gscale*G);
// accumulates ||dW_new||^2 for the next chunk's projection scale.
__global__ __launch_bounds__(256)
void ttt_update(float* __restrict__ dW, const float* __restrict__ G,
                float* __restrict__ scal, const float* __restrict__ log_lr,
                const float* __restrict__ log_dec, int chunkIdx, long perBatch,
                int B) {
  const float eta = __expf(log_lr[0]);
  const float sg = 1.0f / (1.0f + __expf(-log_dec[0]));
  const float decay = 0.9f + (0.995f - 0.9f) * sg;
  const float w0n = sqrtf(scal[0]);

  const long pb4 = perBatch >> 2;
  const long n4 = (long)B * pb4;
  const long stride = (long)gridDim.x * blockDim.x;
  float ss0 = 0.f, ss1 = 0.f;
  v4f* dW4 = (v4f*)dW;
  const v4f* G4 = (const v4f*)G;
  for (long i = (long)blockIdx.x * blockDim.x + threadIdx.x; i < n4; i += stride) {
    const long ip = i + stride;
    if (ip < n4) {                 // stream-ahead prefetch
      __builtin_prefetch(dW4 + ip, 1, 1);
      __builtin_prefetch(G4 + ip, 0, 1);
    }
    const int b = (int)(i / pb4);
    const float gn2 = scal[1 + 2 * chunkIdx + b];
    const float gscale = fminf(G_REL_CAP * w0n / (sqrtf(gn2) + 1e-8f), 1.0f);
    const float dwn2 = scal[33 + 2 * chunkIdx + b];
    const float sprev = fminf(DW_REL_CAP * w0n / (sqrtf(dwn2) + 1e-8f), 1.0f);
    const float ca = decay * sprev;
    const float cg = (1.0f - decay) * eta * gscale;
    v4f w = dW4[i];
    const v4f g = G4[i];
    float d = 0.f;
#pragma unroll
    for (int j = 0; j < 4; ++j) {
      w[j] = ca * w[j] + cg * g[j];
      d += w[j] * w[j];
    }
    dW4[i] = w;
    if (b == 0) ss0 += d; else ss1 += d;
  }
  ss0 = waveReduceSum(ss0);
  ss1 = waveReduceSum(ss1);
  if ((threadIdx.x & 31) == 0) {
    if (ss0 != 0.f) atomicAdd(&scal[33 + 2 * (chunkIdx + 1) + 0], ss0);
    if (ss1 != 0.f) atomicAdd(&scal[33 + 2 * (chunkIdx + 1) + 1], ss1);
  }
}

// ---------------------------------------------------------------------------
extern "C" void kernel_launch(void* const* d_in, const int* in_sizes, int n_in,
                              void* d_out, int out_size, void* d_ws,
                              size_t ws_size, hipStream_t stream) {
  (void)n_in; (void)out_size; (void)ws_size;
  const float* y = (const float*)d_in[0];
  const float* v = (const float*)d_in[1];
  const float* W0 = (const float*)d_in[2];
  const float* bias = (const float*)d_in[3];
  const float* log_lr = (const float*)d_in[4];
  const float* log_dec = (const float*)d_in[5];

  const int d_out_dim = in_sizes[3];
  const int d_in_dim = in_sizes[2] / d_out_dim;
  const long BT = (long)in_sizes[0] / d_in_dim;
  const int B = 2;
  const int T = (int)(BT / B);
  const int C = 256;
  const int nChunks = T / C;
  const long perBatch = (long)d_out_dim * d_in_dim;

  char* ws = (char*)d_ws;
  float* scal = (float*)ws;                         // 128 f32 scalars
  float* dW = (float*)(ws + 512);                   // [B, d_out, d_in] f32
  float* G = dW + (size_t)B * perBatch;             // [B, d_out, d_in] f32
  float* rmsY = G + (size_t)B * perBatch;           // [B*T]
  float* rmsV = rmsY + BT;                          // [B*T]
  half_t* Weff = (half_t*)(rmsV + BT);              // [B, d_out, d_in] f16
  half_t* zf16 = Weff + (size_t)B * perBatch;       // [B, T, d_in] f16
  half_t* znT = zf16 + (size_t)BT * d_in_dim;       // [B, d_in, C] f16
  half_t* vnT = znT + (size_t)B * d_in_dim * C;     // [B, d_out, C] f16

  hipMemsetAsync(scal, 0, 512, stream);
  hipMemsetAsync(dW, 0, sizeof(float) * (size_t)B * perBatch, stream);

  sumsq_kernel<<<512, 256, 0, stream>>>(W0, perBatch / 4, scal);
  row_rms_kernel<<<(int)((BT + 7) / 8), 256, 0, stream>>>(y, rmsY, d_in_dim, BT);
  row_rms_kernel<<<(int)((BT + 7) / 8), 256, 0, stream>>>(v, rmsV, d_out_dim, BT);
  cvt_f16_kernel<<<2048, 256, 0, stream>>>(y, zf16, (BT * d_in_dim) / 8);

  const int gridA = (B * (C / 16) * (d_out_dim / 64)) / 8;
  const int gridG = (B * (d_out_dim / 32) * (d_in_dim / 64)) / 8;
  const int gridTz = B * 4 * (d_in_dim / 64);
  const int gridTv = B * 4 * (d_out_dim / 64);

  float* out = (float*)d_out;
  for (int c = 0; c < nChunks; ++c) {
    prep_weff_kernel<<<2048, 256, 0, stream>>>(W0, dW, scal, Weff, c, perBatch, B);
    transpose_norm_kernel<<<gridTz, 256, 0, stream>>>(y, rmsY, znT, T, d_in_dim, c);
    transpose_norm_kernel<<<gridTv, 256, 0, stream>>>(v, rmsV, vnT, T, d_out_dim, c);
    ttt_apply<<<gridA, 256, 0, stream>>>(zf16, Weff, bias, out, c, T, d_in_dim,
                                         d_out_dim);
    ttt_grad<<<gridG, 256, 0, stream>>>(znT, vnT, G, scal, c, d_in_dim,
                                        d_out_dim);
    ttt_update<<<2048, 256, 0, stream>>>(dW, G, scal, log_lr, log_dec, c,
                                         perBatch, B);
  }
}
